// RelationNet_36069135352015
// MI455X (gfx1250) — compile-verified, matrix operand order checked by reading, not run on hardware
//
#include <hip/hip_runtime.h>
#include <hip/hip_bf16.h>
#include <math.h>

typedef __attribute__((ext_vector_type(16))) __bf16 v16bf;
typedef __attribute__((ext_vector_type(8)))  __bf16 v8bf;
typedef __attribute__((ext_vector_type(8)))  float  v8f;

#define BATCH 256
#define NOBJ  64
#define DF    2048
#define DG    64
#define NR    16
#define DKH   64
#define DVH   128
#define NPROJ (NR*DKH)            // 1024
#define NTOT  (BATCH*NOBJ)        // 16384
#define PAIRS (BATCH*NOBJ*NOBJ)   // 1048576

// ---- Tensor Data Mover availability (compile-safe detection) ---------------
#if defined(__has_builtin)
#  if __has_builtin(__builtin_amdgcn_tensor_load_to_lds) && \
      __has_builtin(__builtin_amdgcn_s_wait_tensorcnt)
#    define USE_TDM 1
#  endif
#endif
#ifndef USE_TDM
#  define USE_TDM 0
#endif
#if __has_include(<hip/amd_detail/amd_gfx1250_TDM.h>)
#  define TDM_SIX_ARGS 1   // therock-10.0 headers -> clang-23 6-arg builtin
#else
#  define TDM_SIX_ARGS 0   // ROCm 7.2 clang-22 5-arg builtin
#endif

__device__ __forceinline__ __bf16 f2bf(float f) {
  union { float f; unsigned u; } x; x.f = f;
  unsigned r = (x.u + 0x7fffu + ((x.u >> 16) & 1u)) >> 16;
  union { unsigned short s; __bf16 b; } y; y.s = (unsigned short)r;
  return y.b;
}

#if USE_TDM
typedef unsigned int tdm_v4u __attribute__((ext_vector_type(4)));
typedef int          tdm_v4i __attribute__((ext_vector_type(4)));
typedef int          tdm_v8i __attribute__((ext_vector_type(8)));

// One TENSOR_LOAD_TO_LDS of a (tile_h x tile_w) bf16 tile from a row-major
// tensor with row stride `stride_elems`. LDS destination gets 16B of padding
// appended after every 128B row (pad_interval=32 dwords, pad_amount=4 dwords)
// => LDS row stride of 72 bf16, matching ld_frag's bank-conflict padding.
__device__ __forceinline__ void tdm_tile(const __bf16* gsrc, __bf16* ldst,
                                         unsigned tile_w, unsigned tile_h,
                                         unsigned stride_elems) {
  unsigned long long ga = (unsigned long long)gsrc;
  tdm_v4u g0;
  g0.x = 1u;                                         // D# count=1, user mode
  g0.y = (unsigned)(unsigned long long)ldst;         // LDS byte offset
  g0.z = (unsigned)ga;                               // global_addr[31:0]
  g0.w = ((unsigned)(ga >> 32) & 0x01FFFFFFu) | (2u << 30);  // [56:32]|type=2
  tdm_v8i g1;
  g1[0] = (int)((1u << 16)      // data_size = 1 (2-byte elements)
              | (1u << 20)      // pad_enable
              | (4u << 22)      // pad_interval: 32 DWORDs (=128B row)
              | (3u << 25));    // pad_amount:    4 DWORDs (=16B)
  g1[1] = (int)((stride_elems & 0xFFFFu) << 16);            // tensor_dim0 lo16
  g1[2] = (int)((stride_elems >> 16) | ((tile_h & 0xFFFFu) << 16)); // d0hi|d1lo
  g1[3] = (int)((tile_h >> 16) | (tile_w << 16));           // d1 hi | tile_dim0
  g1[4] = (int)tile_h;                                      // tile_dim1
  g1[5] = (int)stride_elems;                                // dim0_stride lo32
  g1[6] = 0;
  g1[7] = 0;
  tdm_v4i z4 = {0, 0, 0, 0};
#if TDM_SIX_ARGS
  tdm_v8i z8 = {0, 0, 0, 0, 0, 0, 0, 0};
  __builtin_amdgcn_tensor_load_to_lds(g0, g1, z4, z4, z8, 0);
#else
  __builtin_amdgcn_tensor_load_to_lds(g0, g1, z4, z4, 0);
#endif
}
#endif  // USE_TDM

// Load a 16x32 bf16 WMMA operand fragment from an LDS tile (row-major,
// ld = row stride in elements, multiple of 8 => 16B-aligned b128 loads).
// Lane m (=lane&15), half h (=lane>>4): K elements {h*8..h*8+7, 16+h*8..+7}.
__device__ __forceinline__ v16bf ld_frag(const __bf16* base, int ld, int row0,
                                         int k0, int lane) {
  int m = lane & 15, hf = (lane >> 4) & 1;
  const __bf16* p = base + (row0 + m) * ld + k0 + hf * 8;
  v8bf lo = *(const v8bf*)(p);
  v8bf hi = *(const v8bf*)(p + 16);
  return __builtin_shufflevector(lo, hi, 0,1,2,3,4,5,6,7,8,9,10,11,12,13,14,15);
}

// ----------------------------------------------- fp32 -> bf16 (8 elems/thread)
__global__ void cvt_bf16(const float* __restrict__ in, __bf16* __restrict__ out) {
  long i = ((long)blockIdx.x * 256 + threadIdx.x) * 8;
  float4 a = *(const float4*)(in + i);
  float4 b = *(const float4*)(in + i + 4);
  union { __bf16 h[8]; uint4 u; } pk;
  pk.h[0] = f2bf(a.x); pk.h[1] = f2bf(a.y); pk.h[2] = f2bf(a.z); pk.h[3] = f2bf(a.w);
  pk.h[4] = f2bf(b.x); pk.h[5] = f2bf(b.y); pk.h[6] = f2bf(b.z); pk.h[7] = f2bf(b.w);
  *(uint4*)(out + i) = pk.u;
}

// --------------------------------------------------- log geometric weights
// logg[r][pair] = log(max(g_feature[pair] . WG[r] + bG[r], 1e-6)), head-major
__global__ void __launch_bounds__(256)
geo_logw(const float* __restrict__ g, const float* __restrict__ WG,
         const float* __restrict__ bG, float* __restrict__ logg) {
  __shared__ float wgs[NR][DG];
  __shared__ float bgs[NR];
  int tid = threadIdx.x;
  for (int e = tid; e < NR * DG; e += 256) wgs[e / DG][e % DG] = WG[e];
  if (tid < NR) bgs[tid] = bG[tid];
  __syncthreads();

  long pair = (long)blockIdx.x * 256 + tid;
  const float4* gp = (const float4*)(g + pair * DG);
  float4 gr[DG / 4];
#pragma unroll
  for (int t = 0; t < DG / 4; t++) gr[t] = gp[t];
#pragma unroll
  for (int r = 0; r < NR; r++) {
    float dot = bgs[r];
#pragma unroll
    for (int t = 0; t < DG / 4; t++) {
      dot += gr[t].x * wgs[r][4*t+0] + gr[t].y * wgs[r][4*t+1] +
             gr[t].z * wgs[r][4*t+2] + gr[t].w * wgs[r][4*t+3];
    }
    logg[(long)r * PAIRS + pair] = __logf(fmaxf(dot, 1e-6f));
  }
}

#if !USE_TDM
// Fallback synchronous 64x64 bf16 tile staging (all 128 threads, b128 copies)
__device__ __forceinline__ void stage_copy(const __bf16* Ag, const __bf16* Bg,
                                           __bf16* As, __bf16* Bs, int k0,
                                           int tid) {
#pragma unroll
  for (int t = 0; t < 4; t++) {
    int idx8 = t * 128 + tid;
    int rr = idx8 >> 3, cg = idx8 & 7;
    *(uint4*)(As + rr * 72 + cg * 8) =
        *(const uint4*)(Ag + (long)rr * DF + k0 + cg * 8);
    *(uint4*)(Bs + rr * 72 + cg * 8) =
        *(const uint4*)(Bg + (long)rr * DF + k0 + cg * 8);
  }
}
#endif

// ------------------------------------- K = A_bf16 @ WK_bf16^T + bK  (bf16 out)
// 64x64 block tile, k-step 64 (2 WMMA k-chunks per stage), 4 waves:
// wave w owns output columns w*16..w*16+15.
// Grid x = N-tile (fastest) so the 16 column-blocks of one A row-stripe run
// concurrently and A streams from HBM ~once (stripe + WK stay L2-resident).
// Double-buffered LDS; wave 0 drives the TDM, stage k+1 DMA overlaps stage-k
// WMMAs; s_wait_tensorcnt(2) = "previous stage complete, next in flight".
__global__ void __launch_bounds__(128)
kproj(const __bf16* __restrict__ Ab, const __bf16* __restrict__ Wb,
      const float* __restrict__ bK, __bf16* __restrict__ K) {
  __shared__ __align__(16) __bf16 As[2][64 * 72];
  __shared__ __align__(16) __bf16 Bs[2][64 * 72];
  int tid = threadIdx.x, lane = tid & 31, w = tid >> 5;
  int bn = blockIdx.x, bm = blockIdx.y;
  const int NS = DF / 64;

  const __bf16* Ag = Ab + (long)bm * 64 * DF;
  const __bf16* Bg = Wb + (long)bn * 64 * DF;

  v8f acc[4];
#pragma unroll
  for (int p = 0; p < 4; p++)
#pragma unroll
    for (int q = 0; q < 8; q++) acc[p][q] = 0.f;

  // prologue: stage 0 into buffer 0
#if USE_TDM
  if (w == 0) {
    tdm_tile(Ag, As[0], 64, 64, DF);
    tdm_tile(Bg, Bs[0], 64, 64, DF);
  }
#else
  stage_copy(Ag, Bg, As[0], Bs[0], 0, tid);
#endif

  for (int ks = 0; ks < NS; ks++) {
    int cur = ks & 1;
    if (ks + 1 < NS) {
      int k1 = (ks + 1) * 64;
#if USE_TDM
      if (w == 0) {
        tdm_tile(Ag + k1, As[cur ^ 1], 64, 64, DF);
        tdm_tile(Bg + k1, Bs[cur ^ 1], 64, 64, DF);
        __builtin_amdgcn_s_wait_tensorcnt(2);   // stage ks done, ks+1 in flight
      }
#else
      if (tid < 64) __builtin_prefetch(Ag + (long)tid * DF + k1, 0, 1);
      stage_copy(Ag, Bg, As[cur ^ 1], Bs[cur ^ 1], k1, tid);
#endif
    } else {
#if USE_TDM
      if (w == 0) __builtin_amdgcn_s_wait_tensorcnt(0);
#endif
    }
    __syncthreads();  // stage ks visible to all waves

    const __bf16* Ac = As[cur];
    const __bf16* Bc = Bs[cur];
#pragma unroll
    for (int kk = 0; kk < 2; kk++) {
      v16bf fb  = ld_frag(Bc, 72, w * 16, kk * 32, lane);
      v16bf fa0 = ld_frag(Ac, 72,  0, kk * 32, lane);
      v16bf fa1 = ld_frag(Ac, 72, 16, kk * 32, lane);
      v16bf fa2 = ld_frag(Ac, 72, 32, kk * 32, lane);
      v16bf fa3 = ld_frag(Ac, 72, 48, kk * 32, lane);
      acc[0] = __builtin_amdgcn_wmma_f32_16x16x32_bf16(false, fa0, false, fb,
                                                       (short)0, acc[0], false, false);
      acc[1] = __builtin_amdgcn_wmma_f32_16x16x32_bf16(false, fa1, false, fb,
                                                       (short)0, acc[1], false, false);
      acc[2] = __builtin_amdgcn_wmma_f32_16x16x32_bf16(false, fa2, false, fb,
                                                       (short)0, acc[2], false, false);
      acc[3] = __builtin_amdgcn_wmma_f32_16x16x32_bf16(false, fa3, false, fb,
                                                       (short)0, acc[3], false, false);
    }
    __syncthreads();  // buffer cur released before it is refilled
  }

  int hf = (lane >> 4) & 1, nl = lane & 15;
  int ng = bn * 64 + w * 16 + nl;
  float bias = bK[ng];
#pragma unroll
  for (int ti = 0; ti < 4; ti++)
#pragma unroll
    for (int p = 0; p < 8; p++) {
      int m = bm * 64 + ti * 16 + p + 8 * hf;
      K[(long)m * NPROJ + ng] = f2bf(acc[ti][p] + bias);
    }
}

// ------------------------------------------------- fused attention per (b, r)
__global__ void __launch_bounds__(128)
attn(const __bf16* __restrict__ Abf, const __bf16* __restrict__ Kb,
     const float* __restrict__ logg, const float* __restrict__ swp,
     const float* __restrict__ sbp, float* __restrict__ out) {
  __shared__ __align__(16) __bf16 Kn[64 * 72];   // K head tile  [i][d]
  __shared__ __align__(16) __bf16 Vt[128 * 72];  // V transposed [d][i]
  __shared__ __align__(16) __bf16 Wt[64 * 72];   // w transposed [j][i]
  __shared__ float S[64 * 65];                   // scores + log g, padded
  __shared__ float red[2][64];                   // softmax split-reduction
  int tid = threadIdx.x, lane = tid & 31, w = tid >> 5;
  int b = blockIdx.x, r = blockIdx.y;

  // K head tile: one TDM descriptor (overlaps the Vt staging below)
#if USE_TDM
  if (w == 0)
    tdm_tile(Kb + (long)(b * 64) * NPROJ + r * DKH, Kn, 64, 64, NPROJ);
#else
#pragma unroll
  for (int t = 0; t < 4; t++) {
    int idx8 = t * 128 + tid;
    int i = idx8 >> 3, cg = idx8 & 7;
    *(uint4*)(Kn + i * 72 + cg * 8) =
        *(const uint4*)(Kb + (long)(b * 64 + i) * NPROJ + r * DKH + cg * 8);
  }
#endif
  // V head from bf16 A, transposed into Vt[d][i]: 2 threads/row, 8 b128 each
  {
    int i = tid >> 1, hh = tid & 1;
    const __bf16* src = Abf + (long)(b * 64 + i) * DF + r * DVH + hh * 64;
#pragma unroll
    for (int t = 0; t < 8; t++) {
      union { uint4 u; __bf16 e[8]; } v;
      v.u = *(const uint4*)(src + t * 8);
#pragma unroll
      for (int q = 0; q < 8; q++)
        Vt[(hh * 64 + t * 8 + q) * 72 + i] = v.e[q];
    }
  }
#if USE_TDM
  if (w == 0) __builtin_amdgcn_s_wait_tensorcnt(0);
#endif
  __syncthreads();

  // S = K K^T / 8 + log g ; wave w owns row tile i0 = w*16
  v8f acc[4];
#pragma unroll
  for (int p = 0; p < 4; p++)
#pragma unroll
    for (int q = 0; q < 8; q++) acc[p][q] = 0.f;
#pragma unroll
  for (int ks = 0; ks < 2; ks++) {
    v16bf fa  = ld_frag(Kn, 72, w * 16, ks * 32, lane);
    v16bf fb0 = ld_frag(Kn, 72,  0, ks * 32, lane);   // B = K^T tiles
    v16bf fb1 = ld_frag(Kn, 72, 16, ks * 32, lane);
    v16bf fb2 = ld_frag(Kn, 72, 32, ks * 32, lane);
    v16bf fb3 = ld_frag(Kn, 72, 48, ks * 32, lane);
    acc[0] = __builtin_amdgcn_wmma_f32_16x16x32_bf16(false, fa, false, fb0,
                                                     (short)0, acc[0], false, false);
    acc[1] = __builtin_amdgcn_wmma_f32_16x16x32_bf16(false, fa, false, fb1,
                                                     (short)0, acc[1], false, false);
    acc[2] = __builtin_amdgcn_wmma_f32_16x16x32_bf16(false, fa, false, fb2,
                                                     (short)0, acc[2], false, false);
    acc[3] = __builtin_amdgcn_wmma_f32_16x16x32_bf16(false, fa, false, fb3,
                                                     (short)0, acc[3], false, false);
  }
  int hf = (lane >> 4) & 1, jl = lane & 15;
  const float* lg = logg + (long)r * PAIRS + (long)b * (NOBJ * NOBJ);
#pragma unroll
  for (int tj = 0; tj < 4; tj++)
#pragma unroll
    for (int p = 0; p < 8; p++) {
      int i = w * 16 + p + 8 * hf;
      int j = tj * 16 + jl;
      S[i * 65 + j] = acc[tj][p] * 0.125f + lg[i * 64 + j];
    }
  __syncthreads();

  // softmax over i per column j: 2 threads per column (rows split 32/32)
  {
    int j = tid & 63, hh = tid >> 6;
    int i0 = hh * 32;
    float mx = -1e30f;
    for (int i = i0; i < i0 + 32; i++) mx = fmaxf(mx, S[i * 65 + j]);
    red[hh][j] = mx;
    __syncthreads();
    mx = fmaxf(red[0][j], red[1][j]);
    float sum = 0.f;
    for (int i = i0; i < i0 + 32; i++) sum += __expf(S[i * 65 + j] - mx);
    __syncthreads();
    red[hh][j] = sum;
    __syncthreads();
    float inv = 1.f / (red[0][j] + red[1][j]);
    for (int i = i0; i < i0 + 32; i++)
      Wt[j * 72 + i] = f2bf(__expf(S[i * 65 + j] - mx) * inv);
  }
  __syncthreads();

  // O = w^T @ V ; wave w owns output rows j0 = w*16, 8 d-tiles of 16
  float sw = *swp, sb = *sbp;
  v8f oacc[8];
#pragma unroll
  for (int p = 0; p < 8; p++)
#pragma unroll
    for (int q = 0; q < 8; q++) oacc[p][q] = 0.f;
#pragma unroll
  for (int ks = 0; ks < 2; ks++) {
    v16bf fa = ld_frag(Wt, 72, w * 16, ks * 32, lane);
#pragma unroll
    for (int dt = 0; dt < 8; dt++) {
      v16bf fb = ld_frag(Vt, 72, dt * 16, ks * 32, lane);
      oacc[dt] = __builtin_amdgcn_wmma_f32_16x16x32_bf16(
          false, fa, false, fb, (short)0, oacc[dt], false, false);
    }
  }
#pragma unroll
  for (int dt = 0; dt < 8; dt++)
#pragma unroll
    for (int p = 0; p < 8; p++) {
      int j = w * 16 + p + 8 * hf;
      int d = dt * 16 + jl;
      out[(long)(b * 64 + j) * DF + r * DVH + d] = oacc[dt][p] * sw + sb;
    }
}

extern "C" void kernel_launch(void* const* d_in, const int* in_sizes, int n_in,
                              void* d_out, int out_size, void* d_ws, size_t ws_size,
                              hipStream_t stream) {
  const float* a_feature = (const float*)d_in[0];
  const float* g_feature = (const float*)d_in[1];
  // d_in[2] split (int64), d_in[3] pair_counts (int64): uniform, unused
  const float* WK = (const float*)d_in[4];
  const float* bK = (const float*)d_in[5];
  // d_in[6] WQ, d_in[7] bQ: computed-but-unused in reference forward, skipped
  const float* WG = (const float*)d_in[8];
  const float* bG = (const float*)d_in[9];
  const float* sw = (const float*)d_in[10];
  const float* sb = (const float*)d_in[11];

  char* ws = (char*)d_ws;
  __bf16* abf  = (__bf16*)ws;                                   // 64 MB
  __bf16* wkb  = (__bf16*)(ws + (size_t)67108864);              //  4 MB
  __bf16* kb   = (__bf16*)(ws + (size_t)71303168);              // 32 MB
  float*  logg = (float*)(ws + (size_t)104857600);              // 64 MB

  cvt_bf16<<<(NTOT * DF) / (256 * 8), 256, 0, stream>>>(a_feature, abf);
  cvt_bf16<<<(NPROJ * DF) / (256 * 8), 256, 0, stream>>>(WK, wkb);
  geo_logw<<<PAIRS / 256, 256, 0, stream>>>(g_feature, WG, bG, logg);
  kproj<<<dim3(NPROJ / 64, NTOT / 64), 128, 0, stream>>>(abf, wkb, bK, kb);
  attn<<<dim3(BATCH, NR), 128, 0, stream>>>(abf, kb, logg, sw, sb,
                                            (float*)d_out);
}
